// GLAttention_81776177315856
// MI455X (gfx1250) — matrix operand order repaired and checked by reading, not compiled, not run on hardware
//
#include <hip/hip_runtime.h>
#include <hip/hip_bf16.h>

// ---------------------------------------------------------------------------
// GLAttention fused pipeline for gfx1250 (MI455X), wave32 + WMMA bf16.
//   Wf = w0 @ conv1_w  (64x256), bf = w0 @ conv1_b        [prep]
//   z  = Wf @ x + bf   per batch (64 x 16384)             [gemm0, WMMA]
//   rowmax / rowsumexp over n per (b,k)                   [reduce]
//   p  = exp(z-m)/S ; s[tok]=sum_k p ; fold 1/(eps+s), w2 scalar and
//   b2*rowsum(w1) into epilogue of  out = p^T @ w1^T      [gemm1, WMMA]
// ---------------------------------------------------------------------------

typedef __attribute__((ext_vector_type(16))) __bf16 v16bf;
typedef __attribute__((ext_vector_type(8)))  float  v8f;

#define NTOK 16384
#define CDIM 256
#define KDIM 64
#define BDIM 16

__device__ __forceinline__ unsigned f2bf_u(float f) {
  union { float f; unsigned u; } x; x.f = f;
  return (x.u + 0x7FFFu + ((x.u >> 16) & 1u)) >> 16;   // round-to-nearest-even
}
__device__ __forceinline__ unsigned pack2bf(float lo, float hi) {
  return f2bf_u(lo) | (f2bf_u(hi) << 16);
}

// ---------------------------------------------------------------------------
// prep: blocks 0..63 build Wf row k (bf16) and bf[k]; block 64 builds bf16 w1
// in [c][k] layout (column-major B source) and cbias[c] = b2 * sum_k w1[c][k].
// ---------------------------------------------------------------------------
__global__ void __launch_bounds__(256)
prep_kernel(const float* __restrict__ conv1_w, const float* __restrict__ conv1_b,
            const float* __restrict__ w0, const float* __restrict__ b2,
            const float* __restrict__ w1,
            unsigned* __restrict__ wfbf, float* __restrict__ bfv,
            unsigned* __restrict__ w1bf, float* __restrict__ cbias) {
  int blk = blockIdx.x, tid = threadIdx.x;
  if (blk < KDIM) {
    // thread handles a bf16 pair (two adjacent c) -> one dword store
    int k = blk;
    if (tid < 128) {
      int c = 2 * tid;
      float s0 = 0.f, s1 = 0.f;
      for (int d = 0; d < CDIM; ++d) {
        float w = w0[k * CDIM + d];
        s0 += w * conv1_w[d * CDIM + c];
        s1 += w * conv1_w[d * CDIM + c + 1];
      }
      wfbf[k * 128 + tid] = pack2bf(s0, s1);
    } else if (tid == 128) {
      float sb = 0.f;
      for (int d = 0; d < CDIM; ++d) sb += w0[k * CDIM + d] * conv1_b[d];
      bfv[k] = sb;
    }
  } else {
    int c = tid;
    float s = 0.f;
    for (int j = 0; j < KDIM / 2; ++j) {
      float a = w1[c * KDIM + 2 * j];
      float bb = w1[c * KDIM + 2 * j + 1];
      s += a + bb;
      w1bf[c * 32 + j] = pack2bf(a, bb);
    }
    cbias[c] = b2[0] * s;
  }
}

// ---------------------------------------------------------------------------
// gemm0: z[b][64][16384] = Wf @ x[b] + bf.  Block = (128-token tile, batch).
// Wf (bf16, 32KB) resident in LDS; x K-slabs (32x128) staged transposed and
// double-buffered, packed-dword stores so the ISA 16-bit B layout
// (lane=col, vgpr=K pair) is a contiguous b128 LDS read.
// ---------------------------------------------------------------------------
__global__ void __launch_bounds__(256)
gemm0_kernel(const float* __restrict__ x, const unsigned* __restrict__ wfbf,
             const float* __restrict__ bfv, float* __restrict__ z) {
  __shared__ unsigned sWf[KDIM * 128];     // 64 rows x 128 dwords (256 bf16)
  __shared__ unsigned sB[2][128 * 16];     // ping-pong: 128 cols x 16 dwords
  __shared__ float    sBias[KDIM];

  const int tid = threadIdx.x;
  const int t0  = blockIdx.x * 128;
  const int b   = blockIdx.y;
  const float* xb = x + (size_t)b * CDIM * NTOK;

  for (int i = tid; i < KDIM * 128; i += 256) sWf[i] = wfbf[i];
  if (tid < KDIM) sBias[tid] = bfv[tid];

  const unsigned lane = tid & 31, wave = tid >> 5;
  const unsigned hf = lane >> 4, ln = lane & 15;
  const int mtile = wave & 3;      // 4 M-tiles of 16 (K rows 0..63)
  const int npair = wave >> 2;     // 2 wave groups x 4 N-tiles each

  const int tl  = tid & 127;       // staging: token column
  const int cp0 = (tid >> 7) * 8;  // staging: first of 8 K-pair dwords

  // stage slab 0 into buffer 0
  #pragma unroll
  for (int i = 0; i < 8; ++i) {
    int cp = cp0 + i;
    float v0 = xb[(size_t)(2 * cp)     * NTOK + t0 + tl];
    float v1 = xb[(size_t)(2 * cp + 1) * NTOK + t0 + tl];
    sB[0][tl * 16 + cp] = pack2bf(v0, v1);
  }

  v8f acc[4] = {};
  for (int step = 0; step < 8; ++step) {       // C = 256 in slabs of 32
    __syncthreads();
    if (step < 7) {                            // stage next slab (other buffer)
      #pragma unroll
      for (int i = 0; i < 8; ++i) {
        int cp = cp0 + i;
        float v0 = xb[(size_t)((step + 1) * 32 + 2 * cp)     * NTOK + t0 + tl];
        float v1 = xb[(size_t)((step + 1) * 32 + 2 * cp + 1) * NTOK + t0 + tl];
        sB[(step + 1) & 1][tl * 16 + cp] = pack2bf(v0, v1);
      }
      if (step < 6) {                          // prefetch slab after that
        int e = tid * 16;
        __builtin_prefetch(&xb[(size_t)((step + 2) * 32 + (e >> 7)) * NTOK + t0 + (e & 127)], 0, 3);
      }
    }

    const unsigned* sBc = sB[step & 1];
    union { v16bf v; unsigned u[8]; } af;      // A: 16x32 bf16 fragment
    int arow = mtile * 16 + ln;
    #pragma unroll
    for (int q = 0; q < 8; ++q) {
      int idx = ((q < 4) ? 0 : 8) + 4 * hf + (q & 3);
      af.u[q] = sWf[arow * 128 + step * 16 + idx];
    }
    #pragma unroll
    for (int j = 0; j < 4; ++j) {
      union { v16bf v; unsigned u[8]; } bfr;   // B: 32x16 bf16 fragment
      int col = (npair * 4 + j) * 16 + ln;
      #pragma unroll
      for (int q = 0; q < 8; ++q) bfr.u[q] = sBc[col * 16 + ((q < 4) ? 0 : 8) + 4 * hf + (q & 3)];
      acc[j] = __builtin_amdgcn_wmma_f32_16x16x32_bf16(
          false, af.v, false, bfr.v, (short)0, acc[j], false, false);
    }
  }
  __syncthreads();

  float* zb = z + (size_t)b * KDIM * NTOK;
  #pragma unroll
  for (int j = 0; j < 4; ++j) {
    int col = t0 + (npair * 4 + j) * 16 + ln;
    #pragma unroll
    for (int r = 0; r < 8; ++r) {
      int row = mtile * 16 + r + 8 * hf;       // C/D layout: M = r + 8*(lane>=16)
      zb[(size_t)row * NTOK + col] = acc[j][r] + sBias[row];
    }
  }
}

// ---------------------------------------------------------------------------
// reduce: per (b,k) row of z (16384 f32, L2-resident) -> rowmax, rowsumexp.
// ---------------------------------------------------------------------------
__global__ void __launch_bounds__(256)
softmax_reduce_kernel(const float* __restrict__ z, float* __restrict__ rmax,
                      float* __restrict__ rsum) {
  __shared__ float sred[256];
  const int row = blockIdx.x, tid = threadIdx.x;
  const float* zr = z + (size_t)row * NTOK;
  float m = -3.4e38f;
  for (int i = tid; i < NTOK; i += 256) m = fmaxf(m, zr[i]);
  sred[tid] = m; __syncthreads();
  for (int s = 128; s > 0; s >>= 1) {
    if (tid < s) sred[tid] = fmaxf(sred[tid], sred[tid + s]);
    __syncthreads();
  }
  float rm = sred[0]; __syncthreads();
  float su = 0.f;
  for (int i = tid; i < NTOK; i += 256) su += __expf(zr[i] - rm);
  sred[tid] = su; __syncthreads();
  for (int s = 128; s > 0; s >>= 1) {
    if (tid < s) sred[tid] += sred[tid + s];
    __syncthreads();
  }
  if (tid == 0) { rmax[row] = rm; rsum[row] = sred[0]; }
}

// ---------------------------------------------------------------------------
// gemm1: out[b][tok][c] = w2 * scale[tok] * (p^T @ w1^T) + cbias[c]
// p = exp(z-m)/S computed on the fly; scale = 1/(eps + sum_k p).
// ---------------------------------------------------------------------------
__global__ void __launch_bounds__(256)
gemm1_kernel(const float* __restrict__ z, const float* __restrict__ rmax,
             const float* __restrict__ rsum, const unsigned* __restrict__ w1bf,
             const float* __restrict__ cbias, const float* __restrict__ w2,
             float* __restrict__ out) {
  __shared__ unsigned sW1[CDIM * 32];    // B source: [c][64 bf16] = 32KB
  __shared__ unsigned sA[128 * 32];      // A source: [tok][64 bf16] = 16KB
  __shared__ float sRM[KDIM], sRSi[KDIM], sCB[CDIM], sPart[2][128], sScale[128];

  const int tid = threadIdx.x;
  const int t0  = blockIdx.x * 128;
  const int b   = blockIdx.y;

  for (int i = tid; i < CDIM * 32; i += 256) sW1[i] = w1bf[i];
  if (tid < KDIM) {
    int rr = b * KDIM + tid;
    sRM[tid]  = rmax[rr];
    sRSi[tid] = 1.0f / rsum[rr];
  }
  sCB[tid] = cbias[tid];
  __syncthreads();

  const int token = tid & 127;
  const int hk    = tid >> 7;            // each thread handles 32 K rows
  const float* zb = z + (size_t)b * KDIM * NTOK + t0;
  float sl = 0.f;
  #pragma unroll
  for (int j = 0; j < 16; ++j) {         // 16 packed K-pairs -> b32 stores
    int kk = hk * 32 + 2 * j;
    float p0 = __expf(zb[(size_t)kk       * NTOK + token] - sRM[kk])     * sRSi[kk];
    float p1 = __expf(zb[(size_t)(kk + 1) * NTOK + token] - sRM[kk + 1]) * sRSi[kk + 1];
    sl += p0 + p1;
    sA[token * 32 + hk * 16 + j] = pack2bf(p0, p1);   // A-major layout
  }
  sPart[hk][token] = sl;
  __syncthreads();
  if (tid < 128) sScale[tid] = 1.0f / (1e-9f + sPart[0][tid] + sPart[1][tid]);
  __syncthreads();

  const float ascale = w2[0];
  const unsigned lane = tid & 31, wave = tid >> 5;
  const unsigned hf = lane >> 4, ln = lane & 15;
  const int mtile = wave;                // 8 M-tiles of 16 tokens

  union { v16bf v; unsigned u[8]; } af[2];
  const int arow = mtile * 16 + ln;
  #pragma unroll
  for (int s = 0; s < 2; ++s)
    #pragma unroll
    for (int q = 0; q < 8; ++q) {
      int idx = ((q < 4) ? 0 : 8) + 4 * hf + (q & 3);
      af[s].u[q] = sA[arow * 32 + s * 16 + idx];
    }

  for (int nt = 0; nt < 16; ++nt) {      // 16 N-tiles cover c = 0..255
    v8f acc = {};
    #pragma unroll
    for (int s = 0; s < 2; ++s) {
      union { v16bf v; unsigned u[8]; } bfr;
      int col = nt * 16 + ln;
      #pragma unroll
      for (int q = 0; q < 8; ++q) bfr.u[q] = sW1[col * 32 + s * 16 + ((q < 4) ? 0 : 8) + 4 * hf + (q & 3)];
      acc = __builtin_amdgcn_wmma_f32_16x16x32_bf16(
          false, af[s].v, false, bfr.v, (short)0, acc, false, false);
    }
    #pragma unroll
    for (int r = 0; r < 8; ++r) {
      int ml = mtile * 16 + r + 8 * hf;
      int cc = nt * 16 + ln;
      out[((size_t)b * NTOK + t0 + ml) * CDIM + cc] =
          ascale * sScale[ml] * acc[r] + sCB[cc];
    }
  }
}

// ---------------------------------------------------------------------------
extern "C" void kernel_launch(void* const* d_in, const int* in_sizes, int n_in,
                              void* d_out, int out_size, void* d_ws, size_t ws_size,
                              hipStream_t stream) {
  const float* x       = (const float*)d_in[0];
  const float* conv1_w = (const float*)d_in[1];
  const float* conv1_b = (const float*)d_in[2];
  const float* w0      = (const float*)d_in[3];
  const float* w2      = (const float*)d_in[4];
  const float* b2      = (const float*)d_in[5];
  const float* w1      = (const float*)d_in[6];
  float* out = (float*)d_out;

  char* ws = (char*)d_ws;
  size_t off = 0;
  float*    z     = (float*)(ws + off);  off += (size_t)BDIM * KDIM * NTOK * 4; // 64 MiB
  float*    rmax  = (float*)(ws + off);  off += BDIM * KDIM * 4;
  float*    rsum  = (float*)(ws + off);  off += BDIM * KDIM * 4;
  unsigned* wfbf  = (unsigned*)(ws + off); off += KDIM * CDIM * 2;              // bf16
  unsigned* w1bf  = (unsigned*)(ws + off); off += CDIM * KDIM * 2;              // bf16
  float*    bfv   = (float*)(ws + off);  off += KDIM * 4;
  float*    cbias = (float*)(ws + off);  off += CDIM * 4;

  prep_kernel<<<65, 256, 0, stream>>>(conv1_w, conv1_b, w0, b2, w1,
                                      wfbf, bfv, w1bf, cbias);
  gemm0_kernel<<<dim3(NTOK / 128, BDIM), 256, 0, stream>>>(x, wfbf, bfv, z);
  softmax_reduce_kernel<<<BDIM * KDIM, 256, 0, stream>>>(z, rmax, rsum);
  gemm1_kernel<<<dim3(NTOK / 128, BDIM), 256, 0, stream>>>(z, rmax, rsum, w1bf,
                                                           cbias, w2, out);
}
